// EntropyResidualBlock_80453327388794
// MI455X (gfx1250) — compile-verified, matrix operand order checked by reading, not run on hardware
//
#include <hip/hip_runtime.h>

typedef __bf16 bf16;
typedef __bf16 v16bf __attribute__((ext_vector_type(16)));
typedef __bf16 v8bf  __attribute__((ext_vector_type(8)));
typedef float  v8f   __attribute__((ext_vector_type(8)));

typedef int v4i __attribute__((vector_size(16)));
typedef __attribute__((address_space(1))) v4i v4i_as1;
typedef __attribute__((address_space(3))) v4i v4i_as3;

#define CCH 384      // channels
#define HH  256
#define WW  512
#define CPG 24       // channels per group
#define BM  128      // pixels per block (along W)
#define BN  128      // output channels per block
#define STRIP (BM + 4)
#define CK  32       // ci per K-chunk
#define NCHUNK (CCH / CK)
#define OSTRIDE 68   // staging row stride (64 px + 4 pad floats)

#if defined(__AMDGCN__) && __has_builtin(__builtin_amdgcn_global_load_async_to_lds_b128) && __has_builtin(__builtin_amdgcn_s_wait_asynccnt)
#define HAVE_ASYNC_LDS 1
#else
#define HAVE_ASYNC_LDS 0
#endif

// Build a v16bf A/B fragment for V_WMMA_F32_16X16X32_BF16.
// Per ISA 7.12.2 (16-bit A 16x32): lanes 0-15 hold K 0..7 (VGPR0-3) and
// K 16..23 (VGPR4-7); lanes 16-31 hold K 8..15 and 24..31. With a row-major
// [row][32 ci] LDS layout and per-lane base ci = (lane>>4)*8, that is exactly
// two contiguous 16B loads at +0 and +32B.
static __device__ inline v16bf frag16(const bf16* p) {
  v8bf lo = *(const v8bf*)p;
  v8bf hi = *(const v8bf*)(p + 16);
  v16bf r;
#pragma unroll
  for (int i = 0; i < 8; ++i) { r[i] = lo[i]; r[i + 8] = hi[i]; }
  return r;
}

// Apply PixelCNN group-causal mask, convert fp32 -> bf16, transpose to
// [tap][co][ci] so conv-time B tiles are contiguous 32-ci rows.
__global__ void __launch_bounds__(256)
prep_weights(const float* __restrict__ w1, const float* __restrict__ w2,
             bf16* __restrict__ o1, bf16* __restrict__ o2) {
  const long per = (long)25 * CCH * CCH;
  long idx = (long)blockIdx.x * 256 + threadIdx.x;
  if (idx >= 2 * per) return;
  int layer = idx >= per;
  long e = layer ? idx - per : idx;
  int ci  = (int)(e % CCH);
  int co  = (int)((e / CCH) % CCH);
  int tap = (int)(e / ((long)CCH * CCH));
  int ky = tap / 5, kx = tap % 5;
  float m;
  if (ky < 2)      m = 1.f;
  else if (ky > 2) m = 0.f;
  else             m = (kx < 2) ? 1.f
                       : (kx == 2 ? ((ci / CPG <= co / CPG) ? 1.f : 0.f) : 0.f);
  const float* w = layer ? w2 : w1;
  float v = w[((long)(co * CCH + ci) * 5 + ky) * 5 + kx] * m;
  (layer ? o2 : o1)[e] = (bf16)v;
}

// Implicit-GEMM masked conv + bias + PReLU (+ optional residual).
// Block: BM=128 pixels x BN=128 couts. 8 waves (4 in M x 2 in N), each wave
// 32px x 64co = 2x4 WMMA 16x16 f32 accumulators.
__global__ void __launch_bounds__(256)
entropy_conv(const float* __restrict__ xf, const bf16* __restrict__ xb, int in_bf16,
             const bf16* __restrict__ wt, const float* __restrict__ bias,
             const float* __restrict__ alpha,
             bf16* __restrict__ outb, float* __restrict__ outf,
             const float* __restrict__ resid, int write_f32) {
  extern __shared__ char smem[];
  bf16*  aS = (bf16*)smem;               // STRIP x CK bf16 (8448 B)
  bf16*  bS = aS + STRIP * CK;           // 5 x BN x CK bf16 (40960 B)
  float* oS = (float*)smem;              // BN x OSTRIDE floats (34816 B, aliased)

  const int tid = threadIdx.x;
  const int w0  = blockIdx.x * BM;
  const int h   = blockIdx.y;
  const int bz  = blockIdx.z;
  const int n   = bz / (CCH / BN);
  const int co0 = (bz % (CCH / BN)) * BN;

  const int wv = tid >> 5, lane = tid & 31;
  const int wm = wv & 3, wn = wv >> 2;     // wave tile coords
  const int lr = lane & 15, lh = lane >> 4;
  const int cioff = lh * 8;

  v8f acc[2][4];
#pragma unroll
  for (int mt = 0; mt < 2; ++mt)
#pragma unroll
    for (int nt = 0; nt < 4; ++nt)
#pragma unroll
      for (int i = 0; i < 8; ++i) acc[mt][nt][i] = 0.f;

  for (int ky = 0; ky < 3; ++ky) {              // ky>2 fully masked out
    const int y = h + ky - 2;
    if (y < 0 || y >= HH) continue;             // block-uniform
    const int kxn = (ky == 2) ? 3 : 5;          // center row: kx>2 masked
    for (int cc = 0; cc < NCHUNK; ++cc) {
      const int ci0 = cc * CK;
      // ---- A strip: x[n, ci0..ci0+31, y, w0-2 .. w0+129] -> bf16 LDS [pix][ci]
      for (int idx = tid; idx < CK * STRIP; idx += 256) {
        int ci = idx / STRIP;
        int p  = idx - ci * STRIP;
        int wg = w0 - 2 + p;
        float v = 0.f;
        if (wg >= 0 && wg < WW) {
          size_t off = ((size_t)(n * CCH + ci0 + ci) * HH + y) * WW + wg;
          v = in_bf16 ? (float)xb[off] : xf[off];
          if (cc + 1 < NCHUNK) {                // prefetch next ci-chunk
            size_t nxt = off + (size_t)CK * HH * WW;
            __builtin_prefetch(in_bf16 ? (const void*)(xb + nxt)
                                       : (const void*)(xf + nxt), 0, 1);
          }
        }
        aS[p * CK + ci] = (bf16)v;
      }
      // ---- B tiles: wt[tap][co][ci] raw bf16, 16B per iteration
      for (int idx = tid; idx < kxn * BN * 4; idx += 256) {
        int t  = idx >> 9;                      // tap (kx)
        int r  = idx & 511;
        int co = r >> 2, q = r & 3;
        const bf16* src = wt + (((size_t)(ky * 5 + t) * CCH + (co0 + co)) * CCH + ci0 + q * 8);
        bf16* dst = bS + (t * BN + co) * CK + q * 8;
#if HAVE_ASYNC_LDS
        __builtin_amdgcn_global_load_async_to_lds_b128(
            (v4i_as1*)src, (v4i_as3*)dst, 0, 0);
#else
        *(uint4*)dst = *(const uint4*)src;
#endif
      }
#if HAVE_ASYNC_LDS
      __builtin_amdgcn_s_wait_asynccnt(0);
#endif
      __syncthreads();
      // ---- GEMM: for each live tap, 8 WMMAs per wave
      for (int kx = 0; kx < kxn; ++kx) {
        v16bf af[2], bfg[4];
#pragma unroll
        for (int mt = 0; mt < 2; ++mt) {
          int p = wm * 32 + mt * 16 + lr + kx;  // kx shift = strip offset
          af[mt] = frag16(aS + p * CK + cioff);
        }
#pragma unroll
        for (int nt = 0; nt < 4; ++nt) {
          int co = wn * 64 + nt * 16 + lr;
          bfg[nt] = frag16(bS + (kx * BN + co) * CK + cioff);
        }
#pragma unroll
        for (int mt = 0; mt < 2; ++mt)
#pragma unroll
          for (int nt = 0; nt < 4; ++nt)
            acc[mt][nt] = __builtin_amdgcn_wmma_f32_16x16x32_bf16(
                false, af[mt], false, bfg[nt], (short)0, acc[mt][nt], false, false);
      }
      __syncthreads();
    }
  }

  // ---- Epilogue: transpose through LDS so global stores are coalesced.
  // Two half-passes over pixels (64 each) to keep LDS under the A+B footprint.
  for (int half = 0; half < 2; ++half) {
    if ((wm >> 1) == half) {
#pragma unroll
      for (int mt = 0; mt < 2; ++mt)
#pragma unroll
        for (int nt = 0; nt < 4; ++nt)
#pragma unroll
          for (int r = 0; r < 8; ++r) {
            int coL = wn * 64 + nt * 16 + lr;
            int p   = wm * 32 + mt * 16 + lh * 8 + r - half * 64;
            oS[coL * OSTRIDE + p] = acc[mt][nt][r];
          }
    }
    __syncthreads();
    for (int idx = tid; idx < BN * 16; idx += 256) {
      int co = idx >> 4;
      int p4 = (idx & 15) * 4;
      int cog = co0 + co;
      float bi = bias[cog], al = alpha[cog];
      float4 v = *(float4*)(oS + co * OSTRIDE + p4);
      float r0 = v.x + bi; r0 = r0 >= 0.f ? r0 : al * r0;
      float r1 = v.y + bi; r1 = r1 >= 0.f ? r1 : al * r1;
      float r2 = v.z + bi; r2 = r2 >= 0.f ? r2 : al * r2;
      float r3 = v.w + bi; r3 = r3 >= 0.f ? r3 : al * r3;
      size_t off = ((size_t)(n * CCH + cog) * HH + h) * WW + w0 + half * 64 + p4;
      if (write_f32) {
        float4 rx = *(const float4*)(resid + off);
        float4 o; o.x = r0 + rx.x; o.y = r1 + rx.y; o.z = r2 + rx.z; o.w = r3 + rx.w;
        *(float4*)(outf + off) = o;
      } else {
        struct bf16x4 { bf16 a, b, c, d; };
        struct bf16x4 o = {(bf16)r0, (bf16)r1, (bf16)r2, (bf16)r3};
        *(struct bf16x4*)(outb + off) = o;
      }
    }
    __syncthreads();
  }
}

extern "C" void kernel_launch(void* const* d_in, const int* in_sizes, int n_in,
                              void* d_out, int out_size, void* d_ws, size_t ws_size,
                              hipStream_t stream) {
  (void)in_sizes; (void)n_in; (void)out_size; (void)ws_size;
  const float* x  = (const float*)d_in[0];
  const float* w1 = (const float*)d_in[1];
  const float* b1 = (const float*)d_in[2];
  const float* a1 = (const float*)d_in[3];
  const float* w2 = (const float*)d_in[4];
  const float* b2 = (const float*)d_in[5];
  const float* a2 = (const float*)d_in[6];
  float* out = (float*)d_out;

  const long wE = (long)25 * CCH * CCH;   // 3,686,400 elems per layer
  bf16* wb1 = (bf16*)d_ws;
  bf16* wb2 = wb1 + wE;
  bf16* ybf = wb2 + wE;                   // bf16 intermediate activation

  {
    long total = 2 * wE;
    int blocks = (int)((total + 255) / 256);
    prep_weights<<<blocks, 256, 0, stream>>>(w1, w2, wb1, wb2);
  }

  dim3 grid(WW / BM, HH, 2 * (CCH / BN));            // 4 x 256 x 6
  size_t shmem = (size_t)STRIP * CK * 2 + (size_t)5 * BN * CK * 2;  // 49408 B
  // conv1: fp32 in -> bf16 intermediate (conv + bias + PReLU)
  entropy_conv<<<grid, 256, shmem, stream>>>(x, nullptr, 0, wb1, b1, a1,
                                             ybf, nullptr, nullptr, 0);
  // conv2: bf16 in -> fp32 out (conv + bias + PReLU + residual x)
  entropy_conv<<<grid, 256, shmem, stream>>>(nullptr, ybf, 1, wb2, b2, a2,
                                             nullptr, out, x, 1);
}
